// SEHGN_75368086110469
// MI455X (gfx1250) — matrix operand order repaired and checked by reading, not compiled, not run on hardware
//
#include <hip/hip_runtime.h>
#include <hip/hip_bf16.h>

// ---------------------------------------------------------------------------
// CDNA5 / gfx1250 wave32 WMMA types and fragment helpers
// ---------------------------------------------------------------------------
typedef __attribute__((ext_vector_type(16))) _Float16 v16h;
typedef __attribute__((ext_vector_type(8)))  float    v8f;

union HFrag { v16h h; unsigned int u[8]; };

// A fragment (16x32 f16), source tile stored row-major [rows][lda] halves.
// ISA 7.12.2: lanes 0-15 carry M=0..15 / K pairs {0..7,16..23};
// lanes 16-31 carry the same rows with K pairs {8..15,24..31}.
static __device__ __forceinline__ v16h load_frag_a(const _Float16* s, int lda,
                                                   int row0, int k0) {
  int lane = threadIdx.x & 31;
  int m    = lane & 15;
  int hi   = lane >> 4;
  HFrag f;
  const unsigned int* base = (const unsigned int*)(s + (size_t)(row0 + m) * lda + k0);
#pragma unroll
  for (int j = 0; j < 8; ++j) {
    int k = ((j < 4) ? 0 : 16) + (hi ? 8 : 0) + ((j & 3) << 1);
    f.u[j] = base[k >> 1];
  }
  return f.h;
}

// B fragment (32x16 f16). Source tile stored *transposed* [N][ldb] halves.
// Lanes 0-15: N=lane, K {0..7,16..23} ; lanes 16-31: N=lane-16, K {8..15,24..31}.
static __device__ __forceinline__ v16h load_frag_b(const _Float16* s, int ldb,
                                                   int n0, int k0) {
  int lane = threadIdx.x & 31;
  int n    = lane & 15;
  int hi   = lane >> 4;
  HFrag f;
  const unsigned int* base =
      (const unsigned int*)(s + (size_t)(n0 + n) * ldb + k0);
#pragma unroll
  for (int p = 0; p < 8; ++p) {
    int k = ((p < 4) ? 0 : 16) + (hi ? 8 : 0) + ((p & 3) << 1);
    f.u[p] = base[k >> 1];
  }
  return f.h;
}

static __device__ __forceinline__ v8f wmma_f16(v16h a, v16h b, v8f c) {
  return __builtin_amdgcn_wmma_f32_16x16x32_f16(false, a, false, b,
                                                (short)0, c, false, false);
}

// ---------------------------------------------------------------------------
// Generic tiled WMMA GEMM:  C = act(alpha * opA(A) @ opB(B) + bias [+ Cold])
// A: f32 [M,K] (TRA: physically [K,M], A(m,k)=A[k*lda+m])
// B: f32 [K,N] (TRB: physically [N,K], B(k,n)=B[n*ldb+k])
// Block = 128 threads = 4 wave32 waves; tile 64x64; K step 32.
// ACT: 0 none, 1 tanh, 2 sigmoid.  ACC: C += result.
// All M,N multiples of 64 and K multiples of 32 at every call site.
// ---------------------------------------------------------------------------
template <int TRA, int TRB, int ACT, int ACC>
__global__ __launch_bounds__(128) void gemm_wmma(
    const float* __restrict__ A, const float* __restrict__ B,
    const float* __restrict__ bias, float* __restrict__ C,
    int M, int N, int K, int lda, int ldb, int ldc, float alpha) {
  __shared__ _Float16 As[64 * 32];   // row-major [m][k]
  __shared__ _Float16 Bs[64 * 32];   // transposed [n][k]
  (void)M;
  int tid  = threadIdx.x;
  int wave = tid >> 5;
  int lane = tid & 31;
  int bm = blockIdx.y * 64;
  int bn = blockIdx.x * 64;
  int wm = (wave >> 1) * 32;
  int wn = (wave & 1) * 32;

  v8f acc[2][2] = {};

  for (int k0 = 0; k0 < K; k0 += 32) {
    // stage A 64x32 (f32 -> f16)
#pragma unroll
    for (int i = 0; i < 16; ++i) {
      int idx = i * 128 + tid;          // 0..2047
      int r = idx >> 5, c = idx & 31;   // contiguous K per thread group
      float v = TRA ? A[(size_t)(k0 + c) * lda + (bm + r)]
                    : A[(size_t)(bm + r) * lda + (k0 + c)];
      As[r * 32 + c] = (_Float16)v;
    }
    // stage B 32x64 transposed (f32 -> f16), N fast-varying for coalescing
#pragma unroll
    for (int i = 0; i < 16; ++i) {
      int idx = i * 128 + tid;
      int kk = idx >> 6, n = idx & 63;
      float v = TRB ? B[(size_t)(bn + n) * ldb + (k0 + kk)]
                    : B[(size_t)(k0 + kk) * ldb + (bn + n)];
      Bs[n * 32 + kk] = (_Float16)v;
    }
    if (k0 + 32 < K) {  // hint the next K-panel (global_prefetch_b8)
      __builtin_prefetch(TRA ? &A[(size_t)(k0 + 32) * lda + bm]
                             : &A[(size_t)bm * lda + (k0 + 32)], 0, 1);
      __builtin_prefetch(TRB ? &B[(size_t)bn * ldb + (k0 + 32)]
                             : &B[(size_t)(k0 + 32) * ldb + bn], 0, 1);
    }
    __syncthreads();
#pragma unroll
    for (int ti = 0; ti < 2; ++ti) {
      v16h a = load_frag_a(As, 32, wm + ti * 16, 0);
#pragma unroll
      for (int tj = 0; tj < 2; ++tj) {
        v16h b = load_frag_b(Bs, 32, wn + tj * 16, 0);
        acc[ti][tj] = wmma_f16(a, b, acc[ti][tj]);
      }
    }
    __syncthreads();
  }

  // C/D layout: lanes 0-15 -> N=lane, M=r; lanes 16-31 -> N=lane-16, M=8+r.
  int n_l = lane & 15;
  int hi  = lane >> 4;
#pragma unroll
  for (int ti = 0; ti < 2; ++ti)
#pragma unroll
    for (int tj = 0; tj < 2; ++tj)
#pragma unroll
      for (int r = 0; r < 8; ++r) {
        int row = bm + wm + ti * 16 + (hi ? 8 : 0) + r;
        int col = bn + wn + tj * 16 + n_l;
        float v = alpha * acc[ti][tj][r];
        if (bias != nullptr) v += bias[col];
        if (ACC) v += C[(size_t)row * ldc + col];
        if (ACT == 1) v = tanhf(v);
        else if (ACT == 2) v = 1.0f / (1.0f + __expf(-v));
        C[(size_t)row * ldc + col] = v;
      }
}

// ---------------------------------------------------------------------------
// Fused semantic encoder: one block per description.
// x=emb[des]  [64,256]; per conv c: q=x@Wq, k=x@Wk (WMMA), v=x@Wv [64,4],
// a=softmax(q k^T / 8), o=sigmoid(a v), pooled=max_v, out[l*4+c]=pooled[l].
// LDS: 32K x + 8K q + 8K k + 8K attn + 4K wstage + 2K v/o = 62 KB.
// ---------------------------------------------------------------------------
__global__ __launch_bounds__(128) void semantic_kernel(
    const int* __restrict__ des, const float* __restrict__ emb,
    const float* __restrict__ wq, const float* __restrict__ wk,
    const float* __restrict__ wv, float* __restrict__ out) {
  __shared__ _Float16 x_s[64 * 256];
  __shared__ _Float16 q_s[64 * 64];
  __shared__ _Float16 k_s[64 * 64];
  __shared__ _Float16 a_s[64 * 64];
  __shared__ _Float16 w_s[64 * 32];   // weight tile, transposed [n][k]
  __shared__ float    v_s[64 * 4];
  __shared__ float    o_s[64 * 4];

  int item = blockIdx.x;
  int tid  = threadIdx.x;
  int wave = tid >> 5;
  int lane = tid & 31;
  int n_l  = lane & 15;
  int hi   = lane >> 4;

  // gather token embeddings -> f16 LDS
  for (int i = tid; i < 64 * 256; i += 128) {
    int l = i >> 8, e = i & 255;
    int tok = des[(size_t)item * 64 + l];
    x_s[i] = (_Float16)emb[(size_t)tok * 256 + e];
  }
  __syncthreads();

  for (int c = 0; c < 4; ++c) {
    const float* wqc = wq + (size_t)c * 256 * 64;
    const float* wkc = wk + (size_t)c * 256 * 64;
    const float* wvc = wv + (size_t)c * 256 * 4;

    // ---- q = x@Wq, k = x@Wk  (M=64,N=64,K=256), wave 'w' owns rows 16w..16w+15
    v8f qa[4] = {};
    v8f ka[4] = {};
    for (int k0 = 0; k0 < 256; k0 += 32) {
      for (int i = tid; i < 64 * 32; i += 128) {      // Wq tile -> [n][k]
        int n = i >> 5, kk = i & 31;
        w_s[n * 32 + kk] = (_Float16)wqc[(size_t)(k0 + kk) * 64 + n];
      }
      __syncthreads();
      v16h a = load_frag_a(x_s, 256, wave * 16, k0);
#pragma unroll
      for (int tj = 0; tj < 4; ++tj)
        qa[tj] = wmma_f16(a, load_frag_b(w_s, 32, tj * 16, 0), qa[tj]);
      __syncthreads();
      for (int i = tid; i < 64 * 32; i += 128) {      // Wk tile -> [n][k]
        int n = i >> 5, kk = i & 31;
        w_s[n * 32 + kk] = (_Float16)wkc[(size_t)(k0 + kk) * 64 + n];
      }
      __syncthreads();
#pragma unroll
      for (int tj = 0; tj < 4; ++tj)
        ka[tj] = wmma_f16(a, load_frag_b(w_s, 32, tj * 16, 0), ka[tj]);
      __syncthreads();
    }
#pragma unroll
    for (int tj = 0; tj < 4; ++tj)
#pragma unroll
      for (int r = 0; r < 8; ++r) {
        int row = wave * 16 + (hi ? 8 : 0) + r;
        int col = tj * 16 + n_l;
        q_s[row * 64 + col] = (_Float16)qa[tj][r];
        k_s[row * 64 + col] = (_Float16)ka[tj][r];
      }
    __syncthreads();

    // ---- v = x@Wv  [64,4], plain VALU
    for (int idx = tid; idx < 256; idx += 128) {
      int l = idx >> 2, d = idx & 3;
      float s = 0.f;
      for (int e = 0; e < 256; ++e)
        s += (float)x_s[l * 256 + e] * wvc[(size_t)e * 4 + d];
      v_s[idx] = s;
    }
    __syncthreads();

    // ---- scores = q @ k^T * (1/8): B-transposed layout == k_s as stored
    v8f sc[4] = {};
    for (int k0 = 0; k0 < 64; k0 += 32) {
      v16h a = load_frag_a(q_s, 64, wave * 16, k0);
#pragma unroll
      for (int tj = 0; tj < 4; ++tj)
        sc[tj] = wmma_f16(a, load_frag_b(k_s, 64, tj * 16, k0), sc[tj]);
    }
#pragma unroll
    for (int tj = 0; tj < 4; ++tj)
#pragma unroll
      for (int r = 0; r < 8; ++r) {
        int row = wave * 16 + (hi ? 8 : 0) + r;
        int col = tj * 16 + n_l;
        a_s[row * 64 + col] = (_Float16)(sc[tj][r] * 0.125f);
      }
    __syncthreads();

    // ---- row softmax (64 rows, one thread each)
    if (tid < 64) {
      float mx = -1e30f;
      for (int m = 0; m < 64; ++m) mx = fmaxf(mx, (float)a_s[tid * 64 + m]);
      float sum = 0.f;
      for (int m = 0; m < 64; ++m) {
        float e = __expf((float)a_s[tid * 64 + m] - mx);
        sum += e;
        a_s[tid * 64 + m] = (_Float16)e;
      }
      float inv = 1.0f / sum;
      for (int m = 0; m < 64; ++m)
        a_s[tid * 64 + m] = (_Float16)((float)a_s[tid * 64 + m] * inv);
    }
    __syncthreads();

    // ---- o = sigmoid(a@v) ; pooled = max over v-dim ; write out
    for (int idx = tid; idx < 256; idx += 128) {
      int l = idx >> 2, d = idx & 3;
      float s = 0.f;
      for (int m = 0; m < 64; ++m)
        s += (float)a_s[l * 64 + m] * v_s[m * 4 + d];
      o_s[idx] = 1.0f / (1.0f + __expf(-s));
    }
    __syncthreads();
    if (tid < 64) {
      float mx = o_s[tid * 4];
#pragma unroll
      for (int d = 1; d < 4; ++d) mx = fmaxf(mx, o_s[tid * 4 + d]);
      out[(size_t)item * 256 + tid * 4 + c] = mx;
    }
    __syncthreads();
  }
}

// ---------------------------------------------------------------------------
// Elementwise / sparse helpers
// ---------------------------------------------------------------------------
__global__ void k_zero(float* p, long n) {
  long i = (long)blockIdx.x * blockDim.x + threadIdx.x;
  for (; i < n; i += (long)gridDim.x * blockDim.x) p[i] = 0.0f;
}
__global__ void k_copy(float* d, const float* s, long n, float alpha) {
  long i = (long)blockIdx.x * blockDim.x + threadIdx.x;
  for (; i < n; i += (long)gridDim.x * blockDim.x) d[i] = alpha * s[i];
}
__global__ void k_add(float* d, const float* s, long n) {
  long i = (long)blockIdx.x * blockDim.x + threadIdx.x;
  for (; i < n; i += (long)gridDim.x * blockDim.x) d[i] += s[i];
}
__global__ void k_mul(float* o, const float* a, const float* b, long n) {
  long i = (long)blockIdx.x * blockDim.x + threadIdx.x;
  for (; i < n; i += (long)gridDim.x * blockDim.x) o[i] = a[i] * b[i];
}
// mix = 0.25 * (cw*a + aw*b) / (cw+aw)   (0.25 folds both LightGCN /(K+1))
__global__ void k_mix(float* o, const float* a, const float* b,
                      const float* cw, const float* aw, long n) {
  float c = cw[0], d = aw[0];
  float inv = 0.25f / (c + d);
  long i = (long)blockIdx.x * blockDim.x + threadIdx.x;
  for (; i < n; i += (long)gridDim.x * blockDim.x)
    o[i] = inv * (c * a[i] + d * b[i]);
}
// SpMM scatter: out[rows[e]] += vals[e] * x[cols[e]]   (feature dim 128)
__global__ void k_spmm(const int* __restrict__ rows, const int* __restrict__ cols,
                       const float* __restrict__ vals, const float* __restrict__ x,
                       float* __restrict__ out, int nnz) {
  long t = (long)blockIdx.x * blockDim.x + threadIdx.x;
  int e = (int)(t >> 5);
  int lane = (int)(t & 31);
  if (e < nnz) {
    float v = vals[e];
    int r = rows[e], c = cols[e];
#pragma unroll
    for (int j = 0; j < 4; ++j) {
      int f = lane * 4 + j;
      atomicAdd(&out[(size_t)r * 128 + f], v * x[(size_t)c * 128 + f]);
    }
  }
}
__global__ void k_gather(float* d, const float* s, const int* idx, int width) {
  int b = blockIdx.x;
  for (int j = threadIdx.x; j < width; j += blockDim.x)
    d[(size_t)b * width + j] = s[(size_t)idx[b] * width + j];
}
// L2-normalize each row of P [rows, cols]
__global__ __launch_bounds__(256) void k_rownorm(float* P, int cols) {
  __shared__ float red[256];
  int row = blockIdx.x, tid = threadIdx.x;
  float s = 0.f;
  for (int c = tid; c < cols; c += 256) {
    float v = P[(size_t)row * cols + c];
    s += v * v;
  }
  red[tid] = s;
  __syncthreads();
  for (int off = 128; off; off >>= 1) {
    if (tid < off) red[tid] += red[tid + off];
    __syncthreads();
  }
  float inv = 1.0f / fmaxf(sqrtf(red[0]), 1e-12f);
  for (int c = tid; c < cols; c += 256) P[(size_t)row * cols + c] *= inv;
}

// ---------------------------------------------------------------------------
// Host orchestration
// ---------------------------------------------------------------------------
template <int TRA, int TRB, int ACT, int ACC>
static void launch_gemm(hipStream_t s, const float* A, const float* B,
                        const float* bias, float* C, int M, int N, int K,
                        int lda, int ldb, int ldc, float alpha = 1.0f) {
  dim3 g((unsigned)(N / 64), (unsigned)(M / 64));
  gemm_wmma<TRA, TRB, ACT, ACC><<<g, 128, 0, s>>>(A, B, bias, C, M, N, K,
                                                  lda, ldb, ldc, alpha);
}
static inline dim3 ew_grid(long n) {
  long b = (n + 255) / 256;
  if (b > 65535) b = 65535;
  return dim3((unsigned)b);
}

extern "C" void kernel_launch(void* const* d_in, const int* in_sizes, int n_in,
                              void* d_out, int out_size, void* d_ws,
                              size_t ws_size, hipStream_t stream) {
  (void)n_in; (void)out_size; (void)ws_size;
  const int NM = 8192, NA = 6144, B = 1024, F = 128, E = 256;

  const int*   mashup_des    = (const int*)d_in[0];
  const int*   user_indices  = (const int*)d_in[1];
  const int*   api_des       = (const int*)d_in[2];
  const float* emb_table     = (const float*)d_in[3];
  const float* wq            = (const float*)d_in[4];
  const float* wk            = (const float*)d_in[5];
  const float* wv            = (const float*)d_in[6];
  const float* sc_fcl_w      = (const float*)d_in[7];
  const float* sc_fcl_b      = (const float*)d_in[8];
  const float* api_tag_embed = (const float*)d_in[9];
  const float* tag_w         = (const float*)d_in[10];
  const float* tag_b         = (const float*)d_in[11];
  const float* api_sc_w      = (const float*)d_in[12];
  const float* api_sc_b      = (const float*)d_in[13];
  const float* users_emb     = (const float*)d_in[14];
  const float* items_emb     = (const float*)d_in[15];
  const float* comp_weight   = (const float*)d_in[16];
  const float* affi_weight   = (const float*)d_in[17];
  const float* fic_w         = (const float*)d_in[18];
  const float* fic_b         = (const float*)d_in[19];
  const float* fusion_w      = (const float*)d_in[20];
  const float* fusion_b      = (const float*)d_in[21];
  const float* task_w        = (const float*)d_in[22];
  const float* task_b        = (const float*)d_in[23];
  const int*   ma_rows   = (const int*)d_in[24];
  const int*   ma_cols   = (const int*)d_in[25];
  const float* ma_vals   = (const float*)d_in[26];
  const int*   comp_rows = (const int*)d_in[27];
  const int*   comp_cols = (const int*)d_in[28];
  const float* comp_vals = (const float*)d_in[29];
  const int*   affi_rows = (const int*)d_in[30];
  const int*   affi_cols = (const int*)d_in[31];
  const float* affi_vals = (const float*)d_in[32];
  int nnz_ma   = in_sizes[24];
  int nnz_comp = in_sizes[27];
  int nnz_affi = in_sizes[30];
  float* out = (float*)d_out;

  // workspace layout (f32)
  float* w = (float*)d_ws;
  size_t off = 0;
  auto alloc = [&](size_t n) { float* p = w + off; off += n; return p; };
  float* api_embed  = alloc((size_t)NA * E);
  float* u_embed    = alloc((size_t)B * E);
  float* sc_t       = alloc((size_t)NA * F);
  float* tag_t      = alloc((size_t)NA * F);
  float* api_lin    = alloc((size_t)NA * F);
  float* eb         = alloc((size_t)(NM + NA) * F);
  float* tb         = alloc((size_t)(NM + NA) * F);
  float* accMA      = alloc((size_t)(NM + NA) * F);
  float* e2         = alloc((size_t)2 * NA * F);
  float* t2b        = alloc((size_t)2 * NA * F);
  float* acc2       = alloc((size_t)2 * NA * F);
  float* items_comp = alloc((size_t)NA * F);
  float* mixF       = alloc((size_t)NA * F);
  float* allF       = alloc((size_t)NA * F);
  float* G          = alloc((size_t)F * F);
  float* T2         = alloc((size_t)NM * F);
  float* Hsel       = alloc((size_t)B * F);
  float* u_sc       = alloc((size_t)B * NA);
  float* P          = alloc((size_t)B * NA);
  float* u_mmf      = alloc((size_t)B * NA);

  // 1) semantic encoders (WMMA attention)
  semantic_kernel<<<NA, 128, 0, stream>>>(api_des, emb_table, wq, wk, wv, api_embed);
  semantic_kernel<<<B, 128, 0, stream>>>(mashup_des, emb_table, wq, wk, wv, u_embed);

  // 2) api semantic/tag features
  launch_gemm<0, 0, 1, 0>(stream, api_embed, api_sc_w, api_sc_b, sc_t,
                          NA, F, E, E, F, F);
  launch_gemm<0, 0, 1, 0>(stream, api_tag_embed, tag_w, tag_b, tag_t,
                          NA, F, 2 * E, 2 * E, F, F);
  k_mul<<<ew_grid((long)NA * F), 256, 0, stream>>>(api_lin, sc_t, tag_t, (long)NA * F);

  // 3) u_sc = u_embed @ sc_fcl_w + b
  launch_gemm<0, 0, 0, 0>(stream, u_embed, sc_fcl_w, sc_fcl_b, u_sc,
                          B, NA, E, E, NA, NA);

  // 4) LightGCN propagation (3 hops, scatter-atomic SpMM)
  auto lightgcn = [&](const int* rows, const int* cols, const float* vals, int nnz,
                      const float* top, int ntop, const float* bot, int nbot,
                      float* e, float* t, float* acc) {
    long ntot = (long)(ntop + nbot) * F;
    k_copy<<<ew_grid((long)ntop * F), 256, 0, stream>>>(e, top, (long)ntop * F, 1.0f);
    k_copy<<<ew_grid((long)nbot * F), 256, 0, stream>>>(e + (size_t)ntop * F, bot,
                                                        (long)nbot * F, 1.0f);
    k_copy<<<ew_grid(ntot), 256, 0, stream>>>(acc, e, ntot, 1.0f);
    float* ec = e;
    float* tc = t;
    int blocks = (int)(((long)nnz * 32 + 255) / 256);
    for (int h = 0; h < 3; ++h) {
      k_zero<<<ew_grid(ntot), 256, 0, stream>>>(tc, ntot);
      k_spmm<<<blocks, 256, 0, stream>>>(rows, cols, vals, ec, tc, nnz);
      k_add<<<ew_grid(ntot), 256, 0, stream>>>(acc, tc, ntot);
      float* tmp = ec; ec = tc; tc = tmp;
    }
  };
  lightgcn(ma_rows, ma_cols, ma_vals, nnz_ma, users_emb, NM, items_emb, NA,
           eb, tb, accMA);                                   // accMA = 4*final
  lightgcn(comp_rows, comp_cols, comp_vals, nnz_comp, items_emb, NA, items_emb, NA,
           e2, t2b, acc2);
  k_copy<<<ew_grid((long)NA * F), 256, 0, stream>>>(items_comp, acc2 + (size_t)NA * F,
                                                    (long)NA * F, 1.0f);
  lightgcn(affi_rows, affi_cols, affi_vals, nnz_affi, items_emb, NA, items_emb, NA,
           e2, t2b, acc2);
  // mixF = ((cw*comp + aw*affi)/(cw+aw)) with the /4 LightGCN scale folded in
  k_mix<<<ew_grid((long)NA * F), 256, 0, stream>>>(mixF, items_comp,
                                                   acc2 + (size_t)NA * F,
                                                   comp_weight, affi_weight,
                                                   (long)NA * F);

  // 5) allF = [mixF, api_lin] @ fic_w + fic_b   (split-K: two launches)
  launch_gemm<0, 0, 0, 0>(stream, mixF, fic_w, fic_b, allF, NA, F, F, F, F, F);
  launch_gemm<0, 0, 0, 1>(stream, api_lin, fic_w + (size_t)F * F, nullptr, allF,
                          NA, F, F, F, F, F);

  // 6) low-rank pre: G = all_items^T @ allF ; T2 = all_users @ G ;
  //    pre[u] = T2[u] @ allF^T   (0.25 factors = LightGCN /(K+1))
  launch_gemm<1, 0, 0, 0>(stream, accMA + (size_t)NM * F, allF, nullptr, G,
                          F, F, NA, F, F, F, 0.25f);
  launch_gemm<0, 0, 0, 0>(stream, accMA, G, nullptr, T2,
                          NM, F, F, F, F, F, 0.25f);
  k_gather<<<B, 128, 0, stream>>>(Hsel, T2, user_indices, F);
  launch_gemm<0, 1, 0, 0>(stream, Hsel, allF, nullptr, P, B, NA, F, F, F, NA);
  k_rownorm<<<B, 256, 0, stream>>>(P, NA);   // P -> u_fic

  // 7) fusion: u_mmf = [u_sc, u_fic] @ fusion_w + fusion_b (split-K)
  launch_gemm<0, 0, 0, 0>(stream, u_sc, fusion_w, fusion_b, u_mmf,
                          B, NA, NA, NA, NA, NA);
  launch_gemm<0, 0, 0, 1>(stream, P, fusion_w + (size_t)NA * NA, nullptr, u_mmf,
                          B, NA, NA, NA, NA, NA);

  // 8) task head: out = sigmoid(u_mmf @ task_w + task_b)
  launch_gemm<0, 0, 2, 0>(stream, u_mmf, task_w, task_b, out,
                          B, NA, NA, NA, NA, NA);
}
